// FGS1MambaEncoder_84731114816207
// MI455X (gfx1250) — matrix operand order repaired and checked by reading, not compiled, not run on hardware
//
#include <hip/hip_runtime.h>
#include <hip/hip_bf16.h>

typedef _Float16 half16 __attribute__((ext_vector_type(16)));
typedef float    float8 __attribute__((ext_vector_type(8)));

#define B_SZ    8
#define T_SZ    8192
#define F_IN_SZ 64
#define D_SZ    256
#define L_SZ    6
#define K_CONV  5
#define NCH     32
#define CHUNK   (T_SZ / NCH)   /* 256 */

/* GEMM tiling: 128x64 block tile, 8 waves of 32, each wave a 32x32 tile
 * built from 2x2 v_wmma_f32_16x16x32_f16 fragments. N is always 256. */
#define BM 128
#define BN 64
#define BK 32
#define LDSP 40              /* padded LDS stride in halves (80B rows, 16B aligned) */
#define A_BUF_B (BM * LDSP * 2)   /* 10240 B per A buffer */
#define B_BUF_B (BN * LDSP * 2)   /*  5120 B per B buffer */

// low 32 bits of a flat shared pointer == LDS byte address (aperture in MSBs)
static __device__ __forceinline__ uint32_t lds_lo32(const void* p) {
  return (uint32_t)(uintptr_t)p;
}

// ---------------------------------------------------------------------------
// WMMA GEMM: out[M,256] = epi(A[M,K] @ W[K,256] + bias)
//   A:  f16 row-major [M,K]
//   WT: f16 row-major [256,K]  (pre-transposed weight)
//   EPI 0: out = acc + bias                (in_proj -> h)
//   EPI 1: out = u * sigmoid(acc + bias)   (gate -> u_g)
//   EPI 2: out += acc + bias               (proj residual into h)
// Tiles streamed into double-buffered LDS with global_load_async_to_lds_b128;
// tile k+1 copies overlap tile k WMMA (ASYNCcnt software pipeline).
// ---------------------------------------------------------------------------
template<int EPI>
__global__ __launch_bounds__(256)
void gemm_wmma(const _Float16* __restrict__ A, const _Float16* __restrict__ WT,
               const float* __restrict__ bias, float* __restrict__ out,
               const float* __restrict__ u, int M, int K)
{
  __shared__ _Float16 As[2][BM][LDSP];
  __shared__ _Float16 Bs[2][BN][LDSP];

  const int tid  = threadIdx.x;
  const int lane = tid & 31;
  const int wave = tid >> 5;
  const int wm   = wave & 3;   // m offset 32*wm
  const int wn   = wave >> 2;  // n offset 32*wn
  const int lr   = lane & 15;
  const int lg   = lane >> 4;

  const int m0 = blockIdx.x * BM;
  const int n0 = blockIdx.y * BN;

  const uint32_t asBase = lds_lo32(&As[0][0][0]);
  const uint32_t bsBase = lds_lo32(&Bs[0][0][0]);

  // issue one tile's async copies: 2 A segments + 1 B segment per thread
  auto issue_tile = [&](int kk, int buf) {
    const int k0 = kk * BK;
#pragma unroll
    for (int i = 0; i < 2; ++i) {
      const int s   = tid + 256 * i;       // 0..511
      const int row = s >> 2;              // 4 x 16B segments per 32-half row
      const int seg = s & 3;
      const _Float16* src = A + (size_t)(m0 + row) * K + k0 + seg * 8;
      const uint32_t  dst = asBase + (uint32_t)(buf * A_BUF_B + row * (LDSP * 2) + seg * 16);
      asm volatile("global_load_async_to_lds_b128 %0, %1, off"
                   :: "v"(dst), "v"(src) : "memory");
    }
    {
      const int s   = tid;                 // 0..255
      const int row = s >> 2;
      const int seg = s & 3;
      const _Float16* src = WT + (size_t)(n0 + row) * K + k0 + seg * 8;
      const uint32_t  dst = bsBase + (uint32_t)(buf * B_BUF_B + row * (LDSP * 2) + seg * 16);
      asm volatile("global_load_async_to_lds_b128 %0, %1, off"
                   :: "v"(dst), "v"(src) : "memory");
    }
  };

  const float8 zro = {0.f,0.f,0.f,0.f,0.f,0.f,0.f,0.f};
  float8 acc[2][2];
#pragma unroll
  for (int i = 0; i < 2; ++i)
#pragma unroll
    for (int j = 0; j < 2; ++j) acc[i][j] = zro;

  const int kIters = K / BK;
  issue_tile(0, 0);                        // prologue: fill buffer 0

  for (int kk = 0; kk < kIters; ++kk) {
    const int cur = kk & 1;
    if (kk + 1 < kIters) {
      issue_tile(kk + 1, cur ^ 1);         // stream next tile into other buffer
      if (kk + 2 < kIters) {               // warm L2 for tile k+2
        __builtin_prefetch(A + (size_t)(m0 + (tid >> 1)) * K + (kk + 2) * BK, 0, 1);
        __builtin_prefetch(WT + (size_t)(n0 + (tid & 63)) * K + (kk + 2) * BK, 0, 1);
      }
      // 3 newer ops in flight; asynccnt<=3 => current buffer's 3 ops landed
      asm volatile("s_wait_asynccnt 0x3" ::: "memory");
    } else {
      asm volatile("s_wait_asynccnt 0x0" ::: "memory");
    }
    __syncthreads();

    // ---- fragment gathers per ISA 7.12.2 lane layouts ----
    half16 af[2], bf[2];
#pragma unroll
    for (int fm = 0; fm < 2; ++fm) {
      const int row = wm * 32 + fm * 16 + lr;
#pragma unroll
      for (int v = 0; v < 4; ++v) {                 // VGPR 0..3: K = lg*8 + 2v
        int k = lg * 8 + v * 2;
        af[fm][2 * v]     = As[cur][row][k];
        af[fm][2 * v + 1] = As[cur][row][k + 1];
      }
#pragma unroll
      for (int v = 4; v < 8; ++v) {                 // VGPR 4..7: K = 16 + lg*8 + ...
        int k = 16 + lg * 8 + (v - 4) * 2;
        af[fm][2 * v]     = As[cur][row][k];
        af[fm][2 * v + 1] = As[cur][row][k + 1];
      }
    }
#pragma unroll
    for (int fn = 0; fn < 2; ++fn) {
      const int n = wn * 32 + fn * 16 + lr;
#pragma unroll
      for (int i = 0; i < 8; ++i) {                 // K halves: i<4: lg*8 + 2i, else 16 + lg*8 + 2(i-4)
        int k = (i < 4) ? (lg * 8 + 2 * i) : (16 + lg * 8 + 2 * (i - 4));
        bf[fn][2 * i]     = Bs[cur][n][k];
        bf[fn][2 * i + 1] = Bs[cur][n][k + 1];
      }
    }
#pragma unroll
    for (int fm = 0; fm < 2; ++fm)
#pragma unroll
      for (int fn = 0; fn < 2; ++fn)
        acc[fm][fn] = __builtin_amdgcn_wmma_f32_16x16x32_f16(
            false, af[fm], false, bf[fn], (short)0, acc[fm][fn], false, false);
    __syncthreads();   // all waves done reading buf[cur] before it is refilled
  }

  // ---- epilogue: C/D layout VGPR v -> M = v + 8*lg, N = lr ----
#pragma unroll
  for (int fm = 0; fm < 2; ++fm) {
#pragma unroll
    for (int fn = 0; fn < 2; ++fn) {
      const int ncol = n0 + wn * 32 + fn * 16 + lr;
#pragma unroll
      for (int v = 0; v < 8; ++v) {
        const int token = m0 + wm * 32 + fm * 16 + v + 8 * lg;
        const size_t idx = (size_t)token * D_SZ + ncol;
        float val = acc[fm][fn][v] + bias[ncol];
        if (EPI == 0) {
          out[idx] = val;
        } else if (EPI == 1) {
          float g = 1.0f / (1.0f + __expf(-val));
          out[idx] = u[idx] * g;
        } else {
          out[idx] += val;
        }
      }
    }
  }
}

// ---------------------------------------------------------------------------
// f32 -> f16 elementwise convert (for x)
// ---------------------------------------------------------------------------
__global__ void f16cvt_kernel(const float* __restrict__ in,
                              _Float16* __restrict__ out, int n)
{
  int i = blockIdx.x * 256 + threadIdx.x;
  if (i < n) out[i] = (_Float16)in[i];
}

// Weight transpose+convert: WT[n,k] = (f16) W[k,n].  grid = K, threads = 256.
__global__ __launch_bounds__(256)
void wt_kernel(const float* __restrict__ W, _Float16* __restrict__ WT, int K)
{
  const int k = blockIdx.x, n = threadIdx.x;
  WT[(size_t)n * K + k] = (_Float16)W[(size_t)k * D_SZ + n];
}

// ---------------------------------------------------------------------------
// LayerNorm over D=256, one token per 256-thread block.
// ---------------------------------------------------------------------------
__global__ __launch_bounds__(256)
void ln_kernel(const float* __restrict__ in, const float* __restrict__ w,
               const float* __restrict__ bpar, float* __restrict__ out)
{
  const int token = blockIdx.x, d = threadIdx.x;
  __shared__ float red[256];
  const size_t idx = (size_t)token * D_SZ + d;
  float x = in[idx];
  red[d] = x; __syncthreads();
  for (int s = 128; s > 0; s >>= 1) { if (d < s) red[d] += red[d + s]; __syncthreads(); }
  float mean = red[0] * (1.0f / D_SZ);
  __syncthreads();
  float dx = x - mean;
  red[d] = dx * dx; __syncthreads();
  for (int s = 128; s > 0; s >>= 1) { if (d < s) red[d] += red[d + s]; __syncthreads(); }
  float var = red[0] * (1.0f / D_SZ);
  out[idx] = dx * rsqrtf(var + 1e-5f) * w[d] + bpar[d];
}

// ---------------------------------------------------------------------------
// Depthwise conv1d over T, K=5, 'same' zero padding. Writes f32 (u for the
// gate epilogue) and an f16 mirror (A operand of the gate GEMM).
// ---------------------------------------------------------------------------
__global__ __launch_bounds__(256)
void conv_kernel(const float* __restrict__ z, const float* __restrict__ w,
                 float* __restrict__ z2, _Float16* __restrict__ z2h)
{
  const int blk = blockIdx.x;
  const int b  = blk / (T_SZ / 32);
  const int t0 = (blk % (T_SZ / 32)) * 32;
  const int d  = threadIdx.x;
  float wk[K_CONV];
#pragma unroll
  for (int k = 0; k < K_CONV; ++k) wk[k] = w[d * K_CONV + k];
  for (int tt = 0; tt < 32; ++tt) {
    const int t = t0 + tt;
    float acc = 0.f;
#pragma unroll
    for (int k = 0; k < K_CONV; ++k) {
      const int ts = t + k - 2;
      if (ts >= 0 && ts < T_SZ)
        acc = fmaf(z[((size_t)b * T_SZ + ts) * D_SZ + d], wk[k], acc);
    }
    const size_t idx = ((size_t)b * T_SZ + t) * D_SZ + d;
    z2[idx]  = acc;
    z2h[idx] = (_Float16)acc;
  }
}

__global__ void zero_kernel(float* __restrict__ p, int n)
{
  int i = blockIdx.x * blockDim.x + threadIdx.x;
  if (i < n) p[i] = 0.f;
}

// colsum[b,d] += sum over a 512-token slice of ug  (grid = (B, 16))
__global__ __launch_bounds__(256)
void colsum_kernel(const float* __restrict__ ug, float* __restrict__ colsum)
{
  const int b = blockIdx.x, part = blockIdx.y, d = threadIdx.x;
  const int t0 = part * (T_SZ / 16);
  float s = 0.f;
  for (int t = 0; t < T_SZ / 16; ++t)
    s += ug[((size_t)b * T_SZ + t0 + t) * D_SZ + d];
  atomicAdd(&colsum[b * D_SZ + d], s);
}

// d_eff[b,e] = d[e] + (sum_d u_mean[b,d]*du[d]) * dv[e]   (R = 1)
__global__ __launch_bounds__(256)
void deff_kernel(const float* __restrict__ colsum, const float* __restrict__ du,
                 const float* __restrict__ dv, const float* __restrict__ dpar,
                 float* __restrict__ deff)
{
  const int b = blockIdx.x, d = threadIdx.x;
  __shared__ float red[256];
  float um = colsum[b * D_SZ + d] * (1.0f / T_SZ);
  red[d] = um * du[d];
  __syncthreads();
  for (int s = 128; s > 0; s >>= 1) { if (d < s) red[d] += red[d + s]; __syncthreads(); }
  float alpha = red[0];
  deff[b * D_SZ + d] = dpar[d] + alpha * dv[d];
}

// -------- chunked scan: s[t] = tanh(a)*s[t-1] + b*ug[t] --------
__global__ __launch_bounds__(256)
void scan1_kernel(const float* __restrict__ ug, const float* __restrict__ ap,
                  const float* __restrict__ bp, float* __restrict__ carry)
{
  const int blk = blockIdx.x, b = blk / NCH, ch = blk % NCH, d = threadIdx.x;
  const float at = tanhf(ap[d]), bb = bp[d];
  const size_t base = ((size_t)b * T_SZ + (size_t)ch * CHUNK) * D_SZ + d;
  float s = 0.f;
  for (int i = 0; i < CHUNK; ++i)
    s = fmaf(at, s, bb * ug[base + (size_t)i * D_SZ]);
  carry[((size_t)b * NCH + ch) * D_SZ + d] = s;
}

__global__ __launch_bounds__(256)
void scan2_kernel(const float* __restrict__ carry, const float* __restrict__ ap,
                  float* __restrict__ prefx)
{
  const int b = blockIdx.x, d = threadIdx.x;
  const float at = tanhf(ap[d]);
  const float atC = __powf(at, (float)CHUNK);
  float p = 0.f;
  for (int ch = 0; ch < NCH; ++ch) {
    const size_t idx = ((size_t)b * NCH + ch) * D_SZ + d;
    prefx[idx] = p;
    p = fmaf(atC, p, carry[idx]);
  }
}

// pass 3: replay recurrence with chunk prefix, emit z3 directly in f16
__global__ __launch_bounds__(256)
void scan3_kernel(const float* __restrict__ ug, const float* __restrict__ prefx,
                  const float* __restrict__ ap, const float* __restrict__ bp,
                  const float* __restrict__ cp, const float* __restrict__ deff,
                  _Float16* __restrict__ z3h)
{
  const int blk = blockIdx.x, b = blk / NCH, ch = blk % NCH, d = threadIdx.x;
  const float at = tanhf(ap[d]), bb = bp[d], cc = cp[d];
  const float de = deff[b * D_SZ + d];
  float s = prefx[((size_t)b * NCH + ch) * D_SZ + d];
  const size_t base = ((size_t)b * T_SZ + (size_t)ch * CHUNK) * D_SZ + d;
  for (int i = 0; i < CHUNK; ++i) {
    const float u = ug[base + (size_t)i * D_SZ];
    s = fmaf(at, s, bb * u);
    z3h[base + (size_t)i * D_SZ] = (_Float16)fmaf(cc, s, de * u);
  }
}

// final LayerNorm + mean-pool over T (atomic accumulate, out pre-zeroed)
__global__ __launch_bounds__(256)
void final_kernel(const float* __restrict__ in, const float* __restrict__ w,
                  const float* __restrict__ bpar, float* __restrict__ out)
{
  const int token = blockIdx.x, d = threadIdx.x;
  __shared__ float red[256];
  const size_t idx = (size_t)token * D_SZ + d;
  float x = in[idx];
  red[d] = x; __syncthreads();
  for (int s = 128; s > 0; s >>= 1) { if (d < s) red[d] += red[d + s]; __syncthreads(); }
  float mean = red[0] * (1.0f / D_SZ);
  __syncthreads();
  float dx = x - mean;
  red[d] = dx * dx; __syncthreads();
  for (int s = 128; s > 0; s >>= 1) { if (d < s) red[d] += red[d + s]; __syncthreads(); }
  float var = red[0] * (1.0f / D_SZ);
  float y = dx * rsqrtf(var + 1e-5f) * w[d] + bpar[d];
  const int b = token / T_SZ;
  atomicAdd(&out[b * D_SZ + d], y * (1.0f / T_SZ));
}

// ---------------------------------------------------------------------------
extern "C" void kernel_launch(void* const* d_in, const int* in_sizes, int n_in,
                              void* d_out, int out_size, void* d_ws, size_t ws_size,
                              hipStream_t stream)
{
  const float* x      = (const float*)d_in[0];
  const float* in_w   = (const float*)d_in[1];
  const float* in_b   = (const float*)d_in[2];
  const float* norm_w = (const float*)d_in[3];
  const float* norm_b = (const float*)d_in[4];
  const float* conv_w = (const float*)d_in[5];
  const float* a_p    = (const float*)d_in[6];
  const float* b_p    = (const float*)d_in[7];
  const float* c_p    = (const float*)d_in[8];
  const float* d_p    = (const float*)d_in[9];
  const float* gate_w = (const float*)d_in[10];
  const float* gate_b = (const float*)d_in[11];
  const float* dlr_u  = (const float*)d_in[12];
  const float* dlr_v  = (const float*)d_in[13];
  const float* proj_w = (const float*)d_in[14];
  const float* proj_b = (const float*)d_in[15];
  const float* outn_w = (const float*)d_in[16];
  const float* outn_b = (const float*)d_in[17];
  float* out = (float*)d_out;

  const int    M  = B_SZ * T_SZ;                  // 65536 tokens
  const size_t SZ = (size_t)M * D_SZ;             // 16.7M elements, 64 MiB
  const size_t BD = (size_t)B_SZ * D_SZ;          // 2048
  const size_t BND = (size_t)B_SZ * NCH * D_SZ;   // 65536

  float* h      = (float*)d_ws;                   // persistent residual stream
  float* zug    = h + SZ;                         // LN out, then u_g
  float* z2     = zug + SZ;                       // conv out (u), f32
  float* colsum = z2 + SZ;                        // [B,D]
  float* deff   = colsum + BD;                    // [B,D]
  float* carry  = deff + BD;                      // [B,NCH,D]
  float* prefx  = carry + BND;                    // [B,NCH,D]
  _Float16* ah  = (_Float16*)(prefx + BND);       // f16 activations [M,256]
  _Float16* xh  = ah + SZ;                        // f16 x [M,64]
  _Float16* wh  = xh + (size_t)M * F_IN_SZ;       // f16 W^T [256, K<=256]

  dim3 gblk(256);
  dim3 ggrid(M / BM, D_SZ / BN);                  // 512 x 4

  // x -> f16, W_in^T -> f16, then in_proj: h = x @ in_w + in_b  (K = 64)
  f16cvt_kernel<<<(M * F_IN_SZ + 255) / 256, 256, 0, stream>>>(x, xh, M * F_IN_SZ);
  wt_kernel<<<F_IN_SZ, 256, 0, stream>>>(in_w, wh, F_IN_SZ);
  gemm_wmma<0><<<ggrid, gblk, 0, stream>>>(xh, wh, in_b, h, nullptr, M, F_IN_SZ);

  for (int l = 0; l < L_SZ; ++l) {
    ln_kernel<<<M, 256, 0, stream>>>(h, norm_w + l * D_SZ, norm_b + l * D_SZ, zug);
    conv_kernel<<<B_SZ * (T_SZ / 32), 256, 0, stream>>>(zug, conv_w + l * D_SZ * K_CONV,
                                                        z2, ah);
    // gate: u_g = z2 * sigmoid(z2 @ gate_w + gate_b)  -> zug
    wt_kernel<<<D_SZ, 256, 0, stream>>>(gate_w + (size_t)l * D_SZ * D_SZ, wh, D_SZ);
    gemm_wmma<1><<<ggrid, gblk, 0, stream>>>(ah, wh, gate_b + l * D_SZ, zug, z2, M, D_SZ);
    zero_kernel<<<((int)BD + 255) / 256, 256, 0, stream>>>(colsum, (int)BD);
    colsum_kernel<<<dim3(B_SZ, 16), 256, 0, stream>>>(zug, colsum);
    deff_kernel<<<B_SZ, 256, 0, stream>>>(colsum, dlr_u + l * D_SZ, dlr_v + l * D_SZ,
                                          d_p + l * D_SZ, deff);
    scan1_kernel<<<B_SZ * NCH, 256, 0, stream>>>(zug, a_p + l * D_SZ, b_p + l * D_SZ, carry);
    scan2_kernel<<<B_SZ, 256, 0, stream>>>(carry, a_p + l * D_SZ, prefx);
    scan3_kernel<<<B_SZ * NCH, 256, 0, stream>>>(zug, prefx, a_p + l * D_SZ, b_p + l * D_SZ,
                                                 c_p + l * D_SZ, deff, ah);
    // proj + residual: h += z3 @ proj_w + proj_b
    wt_kernel<<<D_SZ, 256, 0, stream>>>(proj_w + (size_t)l * D_SZ * D_SZ, wh, D_SZ);
    gemm_wmma<2><<<ggrid, gblk, 0, stream>>>(ah, wh, proj_b + l * D_SZ, h, nullptr, M, D_SZ);
  }

  zero_kernel<<<((int)BD + 255) / 256, 256, 0, stream>>>(out, (int)BD);
  final_kernel<<<M, 256, 0, stream>>>(h, outn_w, outn_b, out);
}
